// AttentionModule_16286515986580
// MI455X (gfx1250) — matrix operand order repaired and checked by reading, not compiled, not run on hardware
//
#include <hip/hip_runtime.h>
#include <cstdint>
#include <cstddef>

// ---------------- problem constants ----------------
#define S_LEN 2048
#define D_DIM 4096
#define NH    32
#define NKV   8
#define HDIM  128
#define NREP  (NH / NKV)
#define EPSV  1e-6f

typedef __bf16 bf16;
typedef __attribute__((ext_vector_type(16))) __bf16 bf16x16;
typedef __attribute__((ext_vector_type(8)))  float  f32x8;

// ---------------- bf16 helpers (bit-exact RNE) ----------------
__device__ __forceinline__ unsigned short f2bfu(float x) {
  union { unsigned int u; float f; } in; in.f = x;
  unsigned int r = in.u + 0x7FFFu + ((in.u >> 16) & 1u);
  return (unsigned short)(r >> 16);
}
__device__ __forceinline__ bf16 f2bf(float x) {
  union { unsigned short u; bf16 b; } o; o.u = f2bfu(x); return o.b;
}
__device__ __forceinline__ float bf2f(bf16 x) {
  union { unsigned short u; bf16 b; } in; in.b = x;
  union { unsigned int u; float f; } o; o.u = ((unsigned int)in.u) << 16;
  return o.f;
}

__device__ __forceinline__ void store_out(float* p, float v) { *p = v; }
__device__ __forceinline__ void store_out(bf16*  p, float v) { *p = f2bf(v); }

// ---------------- CDNA5 async global->LDS copy (ISA 15.18.3, op 98) ----------------
// Bypasses VGPRs; tracked with ASYNCcnt. VDST carries the LDS byte address
// (low 32 bits of the flat shared pointer), VADDR the 64-bit global address.
__device__ __forceinline__ void async_load_b128(void* lds_dst, const void* gsrc) {
  unsigned lds_addr = (unsigned)(unsigned long long)lds_dst;  // flat LDS addr[31:0] = byte offset
  asm volatile("global_load_async_to_lds_b128 %0, %1, off"
               :: "v"(lds_addr), "v"(gsrc)
               : "memory");
}
__device__ __forceinline__ void wait_async0() {
  asm volatile("s_wait_asynccnt 0x0" ::: "memory");
}

// ---------------- WMMA fragment loaders (vectorized ds_load_b128) ----------------
// A 16x32 bf16: lane row = lane&15;
// elem j <= K = 16*(lane>>4) + j  (32B contiguous => two 16B loads)
__device__ __forceinline__ bf16x16 load_a_frag(const bf16* Sm, int ld, int rbase, int kbase, int lane) {
  const bf16* p = Sm + (rbase + (lane & 15)) * ld + kbase + ((lane >> 4) << 4);
  union { uint4 v[2]; bf16x16 f; } u;
  u.v[0] = *reinterpret_cast<const uint4*>(p);
  u.v[1] = *reinterpret_cast<const uint4*>(p + 8);
  return u.f;
}
// B 32x16 bf16 stored [col][K] row-major in LDS: lane col = lane&15;
// elem j <= K = 16*(lane>>4) + j  (32B contiguous => two 16B loads)
__device__ __forceinline__ bf16x16 load_b_frag(const bf16* Sm, int ld, int cbase, int kbase, int lane) {
  const bf16* p = Sm + (cbase + (lane & 15)) * ld + kbase + ((lane >> 4) << 4);
  union { uint4 v[2]; bf16x16 f; } u;
  u.v[0] = *reinterpret_cast<const uint4*>(p);
  u.v[1] = *reinterpret_cast<const uint4*>(p + 8);
  return u.f;
}

// ---------------- RMSNorm (f32 in -> bf16 out) ----------------
__global__ __launch_bounds__(256)
void rmsnorm_kernel(const float* __restrict__ x, const float* __restrict__ w,
                    bf16* __restrict__ out) {
  const int row = blockIdx.x;
  const float* xr = x + (size_t)row * D_DIM;
  __shared__ float red[256];
  float ss = 0.f;
  for (int c = threadIdx.x * 4; c < D_DIM; c += 256 * 4) {
    float4 v = *reinterpret_cast<const float4*>(xr + c);
    ss += v.x * v.x + v.y * v.y + v.z * v.z + v.w * v.w;
  }
  red[threadIdx.x] = ss;
  __syncthreads();
  for (int s = 128; s > 0; s >>= 1) {
    if (threadIdx.x < (unsigned)s) red[threadIdx.x] += red[threadIdx.x + s];
    __syncthreads();
  }
  const float inv = rsqrtf(red[0] / (float)D_DIM + EPSV);
  for (int c = threadIdx.x * 4; c < D_DIM; c += 256 * 4) {
    float4 v = *reinterpret_cast<const float4*>(xr + c);
    float4 g = *reinterpret_cast<const float4*>(w + c);
    union { unsigned short us[4]; uint2 u; } pk;
    pk.us[0] = f2bfu(v.x * inv * g.x);
    pk.us[1] = f2bfu(v.y * inv * g.y);
    pk.us[2] = f2bfu(v.z * inv * g.z);
    pk.us[3] = f2bfu(v.w * inv * g.w);
    *reinterpret_cast<uint2*>(&out[(size_t)row * D_DIM + c]) = pk.u;
  }
}

// ---------------- tiled NT GEMM: C[M,N] = A[M,K](bf16) * B[N,K](f32)^T ----------------
// Block tile 64x128, K-step 32, double-buffered LDS, 8 waves as 2x4 grid,
// each wave: 2x2 WMMA tiles (32x32 patch). A-tile staged with async copies.
template <typename OutT>
__global__ __launch_bounds__(256)
void gemm_nt_wmma(const bf16* __restrict__ A, const float* __restrict__ B,
                  OutT* __restrict__ C, int M, int N, int K) {
  constexpr int BM = 64, BN = 128, BK = 32, LDA = BK + 8, LDB = BK + 8;
  __shared__ bf16 As[2][BM * LDA];
  __shared__ bf16 Bs[2][BN * LDB];

  const int tid  = threadIdx.x;
  const int lane = tid & 31;
  const int wave = tid >> 5;
  const int wr   = wave >> 2;   // 0..1
  const int wc   = wave & 3;    // 0..3
  const int m0   = blockIdx.x * BM;
  const int n0   = blockIdx.y * BN;

  const int ar = tid >> 2;          // 0..63
  const int ac = (tid & 3) * 8;     // 0,8,16,24
  const int br = tid >> 1;          // 0..127
  const int bc = (tid & 1) * 16;    // 0,16

  const bf16*  Ag = A + (size_t)(m0 + ar) * K + ac;
  const float* Bg = B + (size_t)(n0 + br) * K + bc;

  f32x8 acc00 = {}, acc01 = {}, acc10 = {}, acc11 = {};

  auto stage = [&](int buf, int k0) {
    // A tile: pure byte copy -> CDNA5 async global->LDS (no VGPR round trip)
    async_load_b128(&As[buf][ar * LDA + ac], Ag + k0);
    // B tile: f32 -> bf16 conversion must go through VGPRs
    const float4* bp = reinterpret_cast<const float4*>(Bg + k0);
    float4 f0 = bp[0], f1 = bp[1], f2 = bp[2], f3 = bp[3];
    union { unsigned short us[8]; uint4 v; } p0, p1;
    p0.us[0] = f2bfu(f0.x); p0.us[1] = f2bfu(f0.y); p0.us[2] = f2bfu(f0.z); p0.us[3] = f2bfu(f0.w);
    p0.us[4] = f2bfu(f1.x); p0.us[5] = f2bfu(f1.y); p0.us[6] = f2bfu(f1.z); p0.us[7] = f2bfu(f1.w);
    p1.us[0] = f2bfu(f2.x); p1.us[1] = f2bfu(f2.y); p1.us[2] = f2bfu(f2.z); p1.us[3] = f2bfu(f2.w);
    p1.us[4] = f2bfu(f3.x); p1.us[5] = f2bfu(f3.y); p1.us[6] = f2bfu(f3.z); p1.us[7] = f2bfu(f3.w);
    *reinterpret_cast<uint4*>(&Bs[buf][br * LDB + bc])     = p0.v;
    *reinterpret_cast<uint4*>(&Bs[buf][br * LDB + bc + 8]) = p1.v;
  };

  stage(0, 0);
  wait_async0();
  __syncthreads();

  const int nk = K / BK;
  for (int kt = 0; kt < nk; ++kt) {
    const int buf = kt & 1;
    if (kt + 1 < nk) stage(buf ^ 1, (kt + 1) * BK);   // async copy overlaps WMMAs below
    if (kt + 2 < nk) {   // gfx1250 speculative prefetch for the K-tile after next
      __builtin_prefetch(Ag + (kt + 2) * BK, 0, 1);
      __builtin_prefetch(Bg + (kt + 2) * BK, 0, 1);
    }
    bf16x16 a0 = load_a_frag(As[buf], LDA, wr * 32,      0, lane);
    bf16x16 a1 = load_a_frag(As[buf], LDA, wr * 32 + 16, 0, lane);
    bf16x16 b0 = load_b_frag(Bs[buf], LDB, wc * 32,      0, lane);
    bf16x16 b1 = load_b_frag(Bs[buf], LDB, wc * 32 + 16, 0, lane);
    acc00 = __builtin_amdgcn_wmma_f32_16x16x32_bf16(false, a0, false, b0, (short)0, acc00, false, false);
    acc01 = __builtin_amdgcn_wmma_f32_16x16x32_bf16(false, a0, false, b1, (short)0, acc01, false, false);
    acc10 = __builtin_amdgcn_wmma_f32_16x16x32_bf16(false, a1, false, b0, (short)0, acc10, false, false);
    acc11 = __builtin_amdgcn_wmma_f32_16x16x32_bf16(false, a1, false, b1, (short)0, acc11, false, false);
    wait_async0();     // async copy for buf^1 must land before the barrier releases readers
    __syncthreads();
  }

  const int rlo = (lane >> 4) << 3;
  const int cl  = lane & 15;
#pragma unroll
  for (int e = 0; e < 8; ++e) {
    const int r0 = m0 + wr * 32 + e + rlo;
    const int r1 = r0 + 16;
    const int c0 = n0 + wc * 32 + cl;
    const int c1 = c0 + 16;
    store_out(&C[(size_t)r0 * N + c0], acc00[e]);
    store_out(&C[(size_t)r0 * N + c1], acc01[e]);
    store_out(&C[(size_t)r1 * N + c0], acc10[e]);
    store_out(&C[(size_t)r1 * N + c1], acc11[e]);
  }
}

// ---------------- RoPE (interleaved pairs, bf16 in/out, f32 math) ----------------
__global__ __launch_bounds__(256)
void rope_kernel(const bf16* __restrict__ in, bf16* __restrict__ out, int nheads) {
  const int idx = blockIdx.x * blockDim.x + threadIdx.x;
  const int total = S_LEN * nheads * (HDIM / 2);
  if (idx >= total) return;
  const int p = idx % (HDIM / 2);
  const int t = idx / (HDIM / 2);     // s*nheads + head
  const int s = t / nheads;
  const float ex   = (float)(2 * p) / (float)HDIM;
  const float freq = powf(50000.0f, -ex);
  const float ang  = (float)s * freq;
  float sn, cs;
  sincosf(ang, &sn, &cs);
  const size_t base = (size_t)t * HDIM + 2 * p;
  union { unsigned short us[2]; unsigned int u; } iv, ov;
  iv.u = *reinterpret_cast<const unsigned int*>(in + base);
  union { unsigned short u; bf16 b; } b0, b1;
  b0.u = iv.us[0]; b1.u = iv.us[1];
  const float x0 = bf2f(b0.b);
  const float x1 = bf2f(b1.b);
  ov.us[0] = f2bfu(x0 * cs - x1 * sn);
  ov.us[1] = f2bfu(x0 * sn + x1 * cs);
  *reinterpret_cast<unsigned int*>(out + base) = ov.u;
}

// ---------------- flash attention (causal, GQA) ----------------
// grid: (S/64 query blocks, NH heads); 256 threads = 8 waves.
// Q tile 64x128 resident (A-fragments hoisted); loop over 32-key tiles;
// K tile staged with async global->LDS copies; V staged TRANSPOSED
// ([dim][key]) so P*V B-fragments are contiguous b128 loads.
__global__ __launch_bounds__(256)
void flash_attn_kernel(const bf16* __restrict__ Q, const bf16* __restrict__ Kt,
                       const bf16* __restrict__ Vt, bf16* __restrict__ O) {
  constexpr int BQ = 64, BKV = 32;
  constexpr int LDQ  = HDIM + 8;   // 136 (272B rows, 16B aligned)
  constexpr int LDV  = BKV + 8;    // 40  (80B rows, 16B aligned)
  constexpr int LDSS = BKV + 4;    // 36
  constexpr int LDP  = BKV + 8;    // 40
  __shared__ bf16  Qs[BQ * LDQ];
  __shared__ bf16  Ks[BKV * LDQ];
  __shared__ bf16  Vs[HDIM * LDV];      // transposed: [dim][key]
  __shared__ float Ss[BQ * LDSS];
  __shared__ bf16  Ps[BQ * LDP];
  __shared__ float rowm[BQ], rowl[BQ], rowsc[BQ];

  const int tid  = threadIdx.x;
  const int lane = tid & 31;
  const int wave = tid >> 5;
  const int qb   = blockIdx.x;
  const int h    = blockIdx.y;
  const int kvh  = h / NREP;
  const int i0   = qb * BQ;
  const float sc_qk = 0.08838834764831843f;   // 1/sqrt(128)

  // ---- stage Q tile once (async byte copy) ----
  {
    const int r  = tid >> 2;
    const int cb = (tid & 3) * 8;
#pragma unroll
    for (int v = 0; v < 4; ++v) {
      const int c = cb + v * 32;
      async_load_b128(&Qs[r * LDQ + c],
                      Q + (size_t)(i0 + r) * (NH * HDIM) + h * HDIM + c);
    }
  }
  if (tid < BQ) { rowm[tid] = -1e30f; rowl[tid] = 0.f; }
  wait_async0();
  __syncthreads();

  const int tr = wave >> 1;   // 0..3: 16-row group
  const int tc = wave & 1;    // 0..1: 16-col score tile / 64-col O half
  const int rlo = (lane >> 4) << 3;
  const int cl  = lane & 15;

  // hoist Q A-fragments (invariant across kv blocks)
  bf16x16 aq[HDIM / 32];
#pragma unroll
  for (int ks = 0; ks < HDIM / 32; ++ks)
    aq[ks] = load_a_frag(Qs, LDQ, tr * 16, ks * 32, lane);

  f32x8 oacc[4] = {};

  const int kvr = tid >> 3;          // 0..31 (key row for staging)
  const int kvc = (tid & 7) * 8;     // 0..56 (dim col for staging)

  const int nkb = 2 * (qb + 1);      // causal: keys up to i0+63
  for (int kbi = 0; kbi < nkb; ++kbi) {
    const int k0 = kbi * BKV;
    __syncthreads();   // prior P*V reads done before restaging K/V/P
    // ---- stage K (async, row-major) and V (VGPR path, transposed) ----
    {
      const size_t kvrow = (size_t)(k0 + kvr) * (NKV * HDIM) + kvh * HDIM;
      async_load_b128(&Ks[kvr * LDQ + kvc],      Kt + kvrow + kvc);
      async_load_b128(&Ks[kvr * LDQ + kvc + 64], Kt + kvrow + kvc + 64);
      union { uint4 v; bf16 e[8]; } v0, v1;
      v0.v = *reinterpret_cast<const uint4*>(Vt + kvrow + kvc);
      v1.v = *reinterpret_cast<const uint4*>(Vt + kvrow + kvc + 64);
#pragma unroll
      for (int i = 0; i < 8; ++i) {
        Vs[(kvc + i) * LDV + kvr]      = v0.e[i];
        Vs[(kvc + 64 + i) * LDV + kvr] = v1.e[i];
      }
      if (kbi + 1 < nkb) {   // speculative prefetch of next kv block
        const size_t nrow = (size_t)(k0 + BKV + kvr) * (NKV * HDIM) + kvh * HDIM;
        __builtin_prefetch(Kt + nrow + kvc, 0, 1);
        __builtin_prefetch(Vt + nrow + kvc, 0, 1);
      }
    }
    wait_async0();
    __syncthreads();

    // ---- scores S = Q*K^T (64x32); wave (tr,tc) owns a 16x16 tile ----
    {
      f32x8 s = {};
#pragma unroll
      for (int ks = 0; ks < HDIM / 32; ++ks) {
        bf16x16 b = load_b_frag(Ks, LDQ, tc * 16, ks * 32, lane);
        s = __builtin_amdgcn_wmma_f32_16x16x32_bf16(false, aq[ks], false, b, (short)0, s, false, false);
      }
#pragma unroll
      for (int e = 0; e < 8; ++e) {
        const int r = tr * 16 + e + rlo;
        const int c = tc * 16 + cl;
        float v = s[e] * sc_qk;
        if (k0 + c > i0 + r) v = -1e30f;   // causal mask
        Ss[r * LDSS + c] = v;
      }
    }
    __syncthreads();

    // ---- online softmax, one thread per query row; packed P stores ----
    if (tid < BQ) {
      const int r = tid;
      const float mprev = rowm[r];
      float sv[BKV];
      float mt = mprev;
#pragma unroll
      for (int c = 0; c < BKV; ++c) { sv[c] = Ss[r * LDSS + c]; mt = fmaxf(mt, sv[c]); }
      const float scale = __expf(mprev - mt);
      float lsum = rowl[r] * scale;
#pragma unroll
      for (int c8 = 0; c8 < BKV / 8; ++c8) {
        union { unsigned short us[8]; uint4 v; } pk;
#pragma unroll
        for (int i = 0; i < 8; ++i) {
          const float p = __expf(sv[c8 * 8 + i] - mt);
          lsum += p;
          pk.us[i] = f2bfu(p);
        }
        *reinterpret_cast<uint4*>(&Ps[r * LDP + c8 * 8]) = pk.v;
      }
      rowm[r] = mt; rowl[r] = lsum; rowsc[r] = scale;
    }
    __syncthreads();

    // ---- rescale O accumulators, then O += P(64x32) * V(32x128) ----
    {
      float sc8[8];
#pragma unroll
      for (int e = 0; e < 8; ++e) sc8[e] = rowsc[tr * 16 + e + rlo];
      bf16x16 a = load_a_frag(Ps, LDP, tr * 16, 0, lane);
#pragma unroll
      for (int j = 0; j < 4; ++j) {
#pragma unroll
        for (int e = 0; e < 8; ++e) oacc[j][e] = oacc[j][e] * sc8[e];
        bf16x16 b = load_b_frag(Vs, LDV, tc * 64 + j * 16, 0, lane);  // contiguous keys
        oacc[j] = __builtin_amdgcn_wmma_f32_16x16x32_bf16(false, a, false, b, (short)0, oacc[j], false, false);
      }
    }
  }
  __syncthreads();

  // ---- epilogue: O *= 1/l ----
  {
    float rinv[8];
#pragma unroll
    for (int e = 0; e < 8; ++e) rinv[e] = 1.0f / rowl[tr * 16 + e + rlo];
#pragma unroll
    for (int j = 0; j < 4; ++j) {
      const int cb = tc * 64 + j * 16;
#pragma unroll
      for (int e = 0; e < 8; ++e) {
        const int r = tr * 16 + e + rlo;
        O[(size_t)(i0 + r) * (NH * HDIM) + h * HDIM + cb + cl] = f2bf(oacc[j][e] * rinv[e]);
      }
    }
  }
}

// ---------------- launcher ----------------
extern "C" void kernel_launch(void* const* d_in, const int* in_sizes, int n_in,
                              void* d_out, int out_size, void* d_ws, size_t ws_size,
                              hipStream_t stream) {
  (void)in_sizes; (void)n_in; (void)out_size; (void)ws_size;
  const float* hidden = (const float*)d_in[0];
  const float* norm_w = (const float*)d_in[1];
  const float* wq     = (const float*)d_in[2];
  const float* wk     = (const float*)d_in[3];
  const float* wv     = (const float*)d_in[4];
  const float* wo     = (const float*)d_in[5];
  float* out = (float*)d_out;

  char* ws = (char*)d_ws;
  size_t off = 0;
  bf16* hb     = (bf16*)(ws + off); off += (size_t)S_LEN * D_DIM * 2;        // 16 MB
  bf16* q_lin  = (bf16*)(ws + off); off += (size_t)S_LEN * NH  * HDIM * 2;   // 16 MB
  bf16* k_lin  = (bf16*)(ws + off); off += (size_t)S_LEN * NKV * HDIM * 2;   //  4 MB
  bf16* v_buf  = (bf16*)(ws + off); off += (size_t)S_LEN * NKV * HDIM * 2;   //  4 MB
  bf16* q_rope = (bf16*)(ws + off); off += (size_t)S_LEN * NH  * HDIM * 2;   // 16 MB
  bf16* k_rope = (bf16*)(ws + off); off += (size_t)S_LEN * NKV * HDIM * 2;   //  4 MB
  bf16* attn   = (bf16*)(ws + off); off += (size_t)S_LEN * NH  * HDIM * 2;   // 16 MB

  // 1) RMSNorm -> bf16 hidden
  rmsnorm_kernel<<<S_LEN, 256, 0, stream>>>(hidden, norm_w, hb);

  // 2) Q/K/V projections (NT gemm, WMMA bf16)
  gemm_nt_wmma<bf16><<<dim3(S_LEN / 64, (NH * HDIM) / 128),  256, 0, stream>>>(hb, wq, q_lin, S_LEN, NH * HDIM,  D_DIM);
  gemm_nt_wmma<bf16><<<dim3(S_LEN / 64, (NKV * HDIM) / 128), 256, 0, stream>>>(hb, wk, k_lin, S_LEN, NKV * HDIM, D_DIM);
  gemm_nt_wmma<bf16><<<dim3(S_LEN / 64, (NKV * HDIM) / 128), 256, 0, stream>>>(hb, wv, v_buf, S_LEN, NKV * HDIM, D_DIM);

  // 3) RoPE on Q and K
  rope_kernel<<<(S_LEN * NH  * (HDIM / 2)) / 256, 256, 0, stream>>>(q_lin, q_rope, NH);
  rope_kernel<<<(S_LEN * NKV * (HDIM / 2)) / 256, 256, 0, stream>>>(k_lin, k_rope, NKV);

  // 4) causal flash attention (GQA via kv-head indexing)
  flash_attn_kernel<<<dim3(S_LEN / 64, NH), 256, 0, stream>>>(q_rope, k_rope, v_buf, attn);

  // 5) output projection -> f32
  gemm_nt_wmma<float><<<dim3(S_LEN / 64, D_DIM / 128), 256, 0, stream>>>(attn, wo, out, S_LEN, D_DIM, D_DIM);
}
